// SelfAttention2d_59124519797011
// MI455X (gfx1250) — matrix operand order repaired and checked by reading, not compiled, not run on hardware
//
#include <hip/hip_runtime.h>
#include <hip/hip_bf16.h>

typedef __bf16 v16bf __attribute__((ext_vector_type(16)));
typedef float  v8f   __attribute__((ext_vector_type(8)));

union frag16 { v16bf v; uint4 q[2]; };

static __device__ __forceinline__ __bf16 f2bf(float f) {
  union { float f; unsigned int u; } a; a.f = f;
  unsigned int r = a.u + 0x7FFFu + ((a.u >> 16) & 1u);   // round-to-nearest-even
  unsigned short h = (unsigned short)(r >> 16);
  return __builtin_bit_cast(__bf16, h);
}

static __device__ __forceinline__ unsigned pack2bf(float lo, float hi) {
  union { float f; unsigned u; } a, b; a.f = lo; b.f = hi;
  unsigned rl = (a.u + 0x7FFFu + ((a.u >> 16) & 1u)) >> 16;
  unsigned rh = (b.u + 0x7FFFu + ((b.u >> 16) & 1u)) >> 16;
  return (rh << 16) | (rl & 0xFFFFu);
}

static __device__ __forceinline__ v8f wmma_bf16(v16bf a, v16bf b, v8f c) {
  // D = A(16x32 bf16) x B(32x16 bf16) + C(16x16 f32)
  return __builtin_amdgcn_wmma_f32_16x16x32_bf16(false, a, false, b, (short)0, c,
                                                 false, false);
}

static __device__ __forceinline__ v8f zero8() {
  v8f z;
#pragma unroll
  for (int i = 0; i < 8; ++i) z[i] = 0.0f;
  return z;
}

// B-fragment (32x16, 16-bit): lane holds col N; half i -> K = 16*g16 + i.
static __device__ __forceinline__ v16bf ld_frag_b(const __bf16* p) {
  frag16 f;
  f.q[0] = *(const uint4*)p;
  f.q[1] = *(const uint4*)(p + 8);
  return f.v;
}

// A-fragment (16x32, 16-bit): lane holds row M; half i -> K = 16*g16 + i.
static __device__ __forceinline__ v16bf ld_frag_a(const __bf16* p) {
  frag16 f;
  f.q[0] = *(const uint4*)p;
  f.q[1] = *(const uint4*)(p + 8);
  return f.v;
}

// gfx1250 async DMA: global -> LDS, 16 bytes/lane, tracked by ASYNCcnt.
static __device__ __forceinline__ void async_load_b128(unsigned lds_addr,
                                                       unsigned byte_off,
                                                       const void* base) {
  asm volatile("global_load_async_to_lds_b128 %0, %1, %2"
               :: "v"(lds_addr), "v"(byte_off), "s"(base) : "memory");
}
static __device__ __forceinline__ void wait_async0() {
  asm volatile("s_wait_asynccnt 0" ::: "memory");
}

// ---------------------------------------------------------------------------
// Kernel 0: convert weights to bf16: W_all = [w_theta; w_phi; w_g] (384x512),
// and w_o (512x256), into workspace.
// ---------------------------------------------------------------------------
__global__ __launch_bounds__(256) void prep_weights(
    const float* __restrict__ wt, const float* __restrict__ wp,
    const float* __restrict__ wg, const float* __restrict__ wo,
    __bf16* __restrict__ wall, __bf16* __restrict__ wobf) {
  int idx = blockIdx.x * 256 + threadIdx.x;
  if (idx < 384 * 512) {
    int row = idx >> 9, k = idx & 511;
    float v;
    if (row < 64)       v = wt[row * 512 + k];
    else if (row < 128) v = wp[(row - 64) * 512 + k];
    else                v = wg[(row - 128) * 512 + k];
    wall[idx] = f2bf(v);
  } else {
    int j = idx - 384 * 512;
    if (j < 512 * 256) wobf[j] = f2bf(wo[j]);
  }
}

// ---------------------------------------------------------------------------
// Kernel 1: projections + fused 2x2 max-pool.
// Grid (32 n-tiles, 3 row-blocks of 128, 16 batches), 256 threads = 8 waves.
// ---------------------------------------------------------------------------
__global__ __launch_bounds__(256) void proj_pool_kernel(
    const float* __restrict__ x, const __bf16* __restrict__ wall,
    __bf16* __restrict__ thetap, __bf16* __restrict__ phip,
    __bf16* __restrict__ gp) {
  __shared__ __attribute__((aligned(16))) __bf16 lds_a[128 * 32];   // [row][k]
  __shared__ __attribute__((aligned(16))) __bf16 lds_b[128 * 32];   // [p][k] (transposed)
  __shared__ __attribute__((aligned(16))) float  lds_c[128 * 128];

  const int tid = threadIdx.x;
  const int lane = tid & 31, w = tid >> 5;
  const int g16 = lane >> 4, l16 = lane & 15;
  const int nt = blockIdx.x, mb = blockIdx.y, b = blockIdx.z;
  const int rb = mb * 128, pbase = nt * 128;

  v8f acc[8];
#pragma unroll
  for (int ni = 0; ni < 8; ++ni) acc[ni] = zero8();

  for (int k0 = 0; k0 < 512; k0 += 32) {
    __syncthreads();
    // A tile [128 x 32] bf16: async DMA straight copy (overlaps x conversion)
    for (int idx = tid; idx < 512; idx += 256) {
      unsigned dst = (unsigned)(uintptr_t)(lds_a + idx * 8);
      unsigned off =
          (unsigned)((((size_t)(rb + (idx >> 2)) * 512) + k0 + (idx & 3) * 8) * 2);
      async_load_b128(dst, off, wall);
    }
    // B tile transposed [128p x 32k]: x fp32 -> packed bf16 pairs (VALU path)
    for (int idx = tid; idx < 2048; idx += 256) {
      int j = idx & 127, kk = (idx >> 7) * 2;
      float f0 = x[(size_t)(b * 512 + k0 + kk) * 4096 + pbase + j];
      float f1 = x[(size_t)(b * 512 + k0 + kk + 1) * 4096 + pbase + j];
      *(unsigned*)&lds_b[j * 32 + kk] = pack2bf(f0, f1);
    }
    wait_async0();
    __syncthreads();

    v16bf af = ld_frag_a(&lds_a[(w * 16 + l16) * 32 + 16 * g16]);
    v16bf bfr = ld_frag_b(&lds_b[l16 * 32 + 16 * g16]);     // ni = 0
#pragma unroll
    for (int ni = 0; ni < 8; ++ni) {
      v16bf cur = bfr;
      if (ni < 7)  // prefetch next B fragment before consuming current
        bfr = ld_frag_b(&lds_b[((ni + 1) * 16 + l16) * 32 + 16 * g16]);
      acc[ni] = wmma_bf16(af, cur, acc[ni]);
    }
  }
  __syncthreads();
  // park D tiles in LDS for the pooling epilogue
#pragma unroll
  for (int ni = 0; ni < 8; ++ni)
#pragma unroll
    for (int r = 0; r < 8; ++r)
      lds_c[(w * 16 + r + 8 * g16) * 128 + ni * 16 + l16] = acc[ni][r];
  __syncthreads();

  if (mb == 0) {  // theta rows 0..63: full resolution
    for (int idx = tid; idx < 64 * 128; idx += 256) {
      int r = idx >> 7, j = idx & 127;
      thetap[((size_t)b * 64 + r) * 4096 + pbase + j] = f2bf(lds_c[idx]);
    }
  }
  // pooled rows (phi / g): a 128-col tile == image rows {2nt,2nt+1}
  for (int idx = tid; idx < 128 * 32; idx += 256) {
    int r = idx >> 5, w2 = idx & 31;
    int gr = rb + r;
    if (gr >= 64) {
      const float* row = lds_c + r * 128;
      float v = fmaxf(fmaxf(row[2 * w2], row[2 * w2 + 1]),
                      fmaxf(row[64 + 2 * w2], row[64 + 2 * w2 + 1]));
      int mcol = nt * 32 + w2;
      if (gr < 128)  // phi stored transposed: [b][m][64k]
        phip[((size_t)b * 1024 + mcol) * 64 + (gr - 64)] = f2bf(v);
      else           // g stored [b][c][1024m]
        gp[((size_t)b * 256 + (gr - 128)) * 1024 + mcol] = f2bf(v);
    }
  }
}

// ---------------------------------------------------------------------------
// Kernel 2: flash attention, double-buffered async tiles.
// Grid (32 q-blocks of 128, 16 batches), 8 waves; wave owns 16 queries.
// DMA of tile mt+1 overlaps WMMA+softmax of tile mt via ASYNCcnt.
// ---------------------------------------------------------------------------
__global__ __launch_bounds__(256) void attn_kernel(
    const __bf16* __restrict__ thetap, const __bf16* __restrict__ phip,
    const __bf16* __restrict__ gp, __bf16* __restrict__ op) {
  __shared__ __attribute__((aligned(16))) __bf16 lds_theta[128 * 64];    // [q][k]
  __shared__ __attribute__((aligned(16))) __bf16 lds_phi[2][64 * 64];    // [m][k]
  __shared__ __attribute__((aligned(16))) __bf16 lds_g[2][256 * 64];     // [c][m]
  __shared__ __attribute__((aligned(16))) __bf16 lds_p[8][16 * 64];      // per-wave P

  const int tid = threadIdx.x;
  const int lane = tid & 31, w = tid >> 5;
  const int g16 = lane >> 4, l16 = lane & 15;
  const int b = blockIdx.y;
  const int qbase = blockIdx.x * 128;

  auto issue_tile = [&](int mt, int buf) {
    for (int idx = tid; idx < 512; idx += 256) {
      unsigned dst = (unsigned)(uintptr_t)(&lds_phi[buf][idx * 8]);
      unsigned off = (unsigned)(((size_t)b * 65536 + (size_t)mt * 4096 + idx * 8) * 2);
      async_load_b128(dst, off, phip);
    }
    for (int idx = tid; idx < 2048; idx += 256) {
      int c = idx >> 3, ch = idx & 7;
      unsigned dst = (unsigned)(uintptr_t)(&lds_g[buf][idx * 8]);
      unsigned off = (unsigned)(((((size_t)b * 256 + c) * 1024) + mt * 64 + ch * 8) * 2);
      async_load_b128(dst, off, gp);
    }
  };

  // stage theta tile transposed [128q x 64k]; kick off tile 0 DMA first
  issue_tile(0, 0);
  for (int idx = tid; idx < 4096; idx += 256) {
    int k = idx >> 6, q2 = (idx & 63) * 2;
    unsigned d = *(const unsigned*)(thetap + ((size_t)b * 64 + k) * 4096 + qbase + q2);
    lds_theta[q2 * 64 + k] = __builtin_bit_cast(__bf16, (unsigned short)(d & 0xFFFFu));
    lds_theta[(q2 + 1) * 64 + k] = __builtin_bit_cast(__bf16, (unsigned short)(d >> 16));
  }
  __syncthreads();

  // theta A-fragments (A = theta^T : [16q x 64k]), registers for all m
  v16bf Ath[2];
#pragma unroll
  for (int kt = 0; kt < 2; ++kt)
    Ath[kt] = ld_frag_a(&lds_theta[(w * 16 + l16) * 64 + kt * 32 + 16 * g16]);

  v8f O[16];
#pragma unroll
  for (int ci = 0; ci < 16; ++ci) O[ci] = zero8();
  v8f run_max, run_sum;
#pragma unroll
  for (int r = 0; r < 8; ++r) { run_max[r] = -__builtin_inff(); run_sum[r] = 0.0f; }

  for (int mt = 0; mt < 16; ++mt) {
    const int cur = mt & 1;
    wait_async0();        // tile mt resident (own wave's DMAs)
    __syncthreads();      // ... and everyone else's; prior reads of buf^1 done
    if (mt + 1 < 16) issue_tile(mt + 1, cur ^ 1);  // overlap with compute below

    const __bf16* phi_t = lds_phi[cur];
    const __bf16* g_t = lds_g[cur];

    // S[16q x 64m] : 4 sub-tiles of 16 columns (B-frag loads pipelined)
    v8f S[4];
    {
      v16bf bfr = ld_frag_b(&phi_t[l16 * 64 + 16 * g16]);  // ni=0, kt=0
#pragma unroll
      for (int ni = 0; ni < 4; ++ni) {
        v8f s = zero8();
#pragma unroll
        for (int kt = 0; kt < 2; ++kt) {
          v16bf curf = bfr;
          int j = ni * 2 + kt + 1;
          if (j < 8) {
            int nni = j >> 1, nkt = j & 1;
            bfr = ld_frag_b(&phi_t[(nni * 16 + l16) * 64 + nkt * 32 + 16 * g16]);
          }
          s = wmma_bf16(Ath[kt], curf, s);
        }
        S[ni] = s;
      }
    }

    // online softmax: row groups match D-tile halves (16-lane butterflies)
    v8f tmax;
#pragma unroll
    for (int r = 0; r < 8; ++r) {
      float v = fmaxf(fmaxf(S[0][r], S[1][r]), fmaxf(S[2][r], S[3][r]));
      v = fmaxf(v, __shfl_xor(v, 1, 32));
      v = fmaxf(v, __shfl_xor(v, 2, 32));
      v = fmaxf(v, __shfl_xor(v, 4, 32));
      v = fmaxf(v, __shfl_xor(v, 8, 32));
      tmax[r] = v;
    }
    v8f scale, psum;
#pragma unroll
    for (int r = 0; r < 8; ++r) {
      float nm = fmaxf(run_max[r], tmax[r]);
      scale[r] = __expf(run_max[r] - nm);
      run_max[r] = nm;
      psum[r] = 0.0f;
    }
#pragma unroll
    for (int ni = 0; ni < 4; ++ni)
#pragma unroll
      for (int r = 0; r < 8; ++r) {
        float p = __expf(S[ni][r] - run_max[r]);
        S[ni][r] = p;
        psum[r] += p;
      }
#pragma unroll
    for (int r = 0; r < 8; ++r) {
      float v = psum[r];
      v += __shfl_xor(v, 1, 32);
      v += __shfl_xor(v, 2, 32);
      v += __shfl_xor(v, 4, 32);
      v += __shfl_xor(v, 8, 32);
      run_sum[r] = run_sum[r] * scale[r] + v;
    }
#pragma unroll
    for (int ci = 0; ci < 16; ++ci)
#pragma unroll
      for (int r = 0; r < 8; ++r) O[ci][r] *= scale[r];

    // P: D-layout -> A-layout via per-wave LDS (wave-local, DS is in-order)
#pragma unroll
    for (int ni = 0; ni < 4; ++ni)
#pragma unroll
      for (int r = 0; r < 8; ++r)
        lds_p[w][(r + 8 * g16) * 64 + ni * 16 + l16] = f2bf(S[ni][r]);
    asm volatile("s_wait_dscnt 0" ::: "memory");

    v16bf Pf[2];
#pragma unroll
    for (int kt = 0; kt < 2; ++kt)
      Pf[kt] = ld_frag_a(&lds_p[w][l16 * 64 + kt * 32 + 16 * g16]);

    // O[16q x 256c] += P[16q x 64m] x g^T[64m x 256c]  (pipelined B-frags)
    {
      v16bf bfr = ld_frag_b(&g_t[l16 * 64 + 16 * g16]);  // ci=0, kt=0
#pragma unroll
      for (int j = 0; j < 32; ++j) {
        int ci = j >> 1, kt = j & 1;
        v16bf curf = bfr;
        if (j < 31) {
          int nci = (j + 1) >> 1, nkt = (j + 1) & 1;
          bfr = ld_frag_b(&g_t[(nci * 16 + l16) * 64 + nkt * 32 + 16 * g16]);
        }
        O[ci] = wmma_bf16(Pf[kt], curf, O[ci]);
      }
    }
  }

  // normalize and store o[b][q][256c] as bf16
  v8f rinv;
#pragma unroll
  for (int r = 0; r < 8; ++r) rinv[r] = 1.0f / run_sum[r];
#pragma unroll
  for (int ci = 0; ci < 16; ++ci)
#pragma unroll
    for (int r = 0; r < 8; ++r) {
      int q = qbase + w * 16 + r + 8 * g16;
      int c = ci * 16 + l16;
      op[((size_t)b * 4096 + q) * 256 + c] = f2bf(O[ci][r] * rinv[r]);
    }
}

// ---------------------------------------------------------------------------
// Kernel 3: out = gamma * (w_o x o) + x. Grid (32 n-tiles, 4 row-blocks, 16 b).
// ---------------------------------------------------------------------------
__global__ __launch_bounds__(256) void out_proj_kernel(
    const float* __restrict__ x, const __bf16* __restrict__ wobf,
    const __bf16* __restrict__ op, const float* __restrict__ gammap,
    float* __restrict__ out) {
  __shared__ __attribute__((aligned(16))) __bf16 lds_a[128 * 32];   // [row][k]
  __shared__ __attribute__((aligned(16))) __bf16 lds_b[128 * 32];   // [p][k] (transposed)

  const int tid = threadIdx.x;
  const int lane = tid & 31, w = tid >> 5;
  const int g16 = lane >> 4, l16 = lane & 15;
  const int nt = blockIdx.x, mb = blockIdx.y, b = blockIdx.z;
  const int rb = mb * 128, pbase = nt * 128;

  v8f acc[8];
#pragma unroll
  for (int ni = 0; ni < 8; ++ni) acc[ni] = zero8();

  for (int k0 = 0; k0 < 256; k0 += 32) {
    __syncthreads();
    // both tiles are straight bf16 copies -> async DMA to LDS
    for (int idx = tid; idx < 512; idx += 256) {
      unsigned dst = (unsigned)(uintptr_t)(lds_a + idx * 8);
      unsigned off =
          (unsigned)((((size_t)(rb + (idx >> 2)) * 256) + k0 + (idx & 3) * 8) * 2);
      async_load_b128(dst, off, wobf);
    }
    for (int idx = tid; idx < 512; idx += 256) {
      int p = idx >> 2, ch = idx & 3;
      unsigned dst = (unsigned)(uintptr_t)(lds_b + idx * 8);
      unsigned off =
          (unsigned)(((((size_t)b * 4096 + pbase + p) * 256) + k0 + ch * 8) * 2);
      async_load_b128(dst, off, op);
    }
    wait_async0();
    __syncthreads();

    v16bf af = ld_frag_a(&lds_a[(w * 16 + l16) * 32 + 16 * g16]);
    v16bf bfr = ld_frag_b(&lds_b[l16 * 32 + 16 * g16]);  // ni = 0
#pragma unroll
    for (int ni = 0; ni < 8; ++ni) {
      v16bf cur = bfr;
      if (ni < 7)
        bfr = ld_frag_b(&lds_b[((ni + 1) * 16 + l16) * 32 + 16 * g16]);
      acc[ni] = wmma_bf16(af, cur, acc[ni]);
    }
  }

  float gamma = gammap[0];
#pragma unroll
  for (int ni = 0; ni < 8; ++ni)
#pragma unroll
    for (int r = 0; r < 8; ++r) {
      int och = rb + w * 16 + r + 8 * g16;
      int p = pbase + ni * 16 + l16;
      size_t idx = ((size_t)b * 512 + och) * 4096 + p;
      out[idx] = gamma * acc[ni][r] + x[idx];
    }
}

// ---------------------------------------------------------------------------
extern "C" void kernel_launch(void* const* d_in, const int* in_sizes, int n_in,
                              void* d_out, int out_size, void* d_ws, size_t ws_size,
                              hipStream_t stream) {
  (void)in_sizes; (void)n_in; (void)out_size; (void)ws_size;
  const float* x     = (const float*)d_in[0];
  const float* wt    = (const float*)d_in[1];
  const float* wp    = (const float*)d_in[2];
  const float* wg    = (const float*)d_in[3];
  const float* wo    = (const float*)d_in[4];
  const float* gamma = (const float*)d_in[5];
  float* out = (float*)d_out;

  // workspace layout (bf16 elements), total ~53 MB
  __bf16* wall   = (__bf16*)d_ws;                        // 384*512
  __bf16* wobf   = wall + 384 * 512;                     // 512*256
  __bf16* thetap = wobf + 512 * 256;                     // [16][64][4096]
  __bf16* phip   = thetap + (size_t)16 * 64 * 4096;      // [16][1024][64]
  __bf16* gp     = phip + (size_t)16 * 1024 * 64;        // [16][256][1024]
  __bf16* op     = gp + (size_t)16 * 256 * 1024;         // [16][4096][256]

  prep_weights<<<1280, 256, 0, stream>>>(wt, wp, wg, wo, wall, wobf);
  proj_pool_kernel<<<dim3(32, 3, 16), 256, 0, stream>>>(x, wall, thetap, phip, gp);
  attn_kernel<<<dim3(32, 16), 256, 0, stream>>>(thetap, phip, gp, op);
  out_proj_kernel<<<dim3(32, 4, 16), 256, 0, stream>>>(x, wobf, op, gamma, out);
}
